// Encoder_1030792151055
// MI455X (gfx1250) — matrix operand order repaired and checked by reading, not compiled, not run on hardware
//
#include <hip/hip_runtime.h>

#define BOARD 361
#define SEQL  24
#define B_TOT 4096
#define H     256
#define G4    1024   // 4*H gate width
#define BT    32     // batch tile per workgroup
#define PAD   272    // padded LDS row length (elements), 16B-aligned rows

typedef __attribute__((ext_vector_type(8)))  __bf16 v8bf;
typedef __attribute__((ext_vector_type(16))) __bf16 v16bf;
typedef __attribute__((ext_vector_type(8)))  float  v8f;

// ---------------- packing kernels ----------------

__global__ void k_f32_to_bf16(const float* __restrict__ src, __bf16* __restrict__ dst, int n) {
    int i = blockIdx.x * blockDim.x + threadIdx.x;
    if (i < n) dst[i] = (__bf16)src[i];
}

// src [2][1024][361] -> dst [2][361][1024]  (so per-position gather is contiguous over gates)
__global__ void k_transpose_wih0(const float* __restrict__ src, float* __restrict__ dst) {
    int i = blockIdx.x * blockDim.x + threadIdx.x;
    int total = 2 * BOARD * G4;
    if (i >= total) return;
    int n = i % G4;
    int p = (i / G4) % BOARD;
    int d = i / (G4 * BOARD);
    dst[i] = src[(size_t)d * G4 * BOARD + (size_t)n * BOARD + p];
}

// bias[cell][n] = b_ih + b_hh, cell = layer*2 + dir
__global__ void k_bias(const float* __restrict__ bih0, const float* __restrict__ bhh0,
                       const float* __restrict__ bih1, const float* __restrict__ bhh1,
                       float* __restrict__ dst) {
    int i = blockIdx.x * blockDim.x + threadIdx.x;
    if (i >= 4 * G4) return;
    int cell = i / G4, n = i % G4;
    int l = cell >> 1, d = cell & 1;
    const float* bi = l ? bih1 : bih0;
    const float* bh = l ? bhh1 : bhh0;
    dst[i] = bi[d * G4 + n] + bh[d * G4 + n];
}

// ---------------- main LSTM kernel ----------------

__device__ __forceinline__ float sigm(float x) { return 1.0f / (1.0f + __expf(-x)); }
__device__ __forceinline__ float ftanh(float x) {
    float e = __expf(2.0f * x);
    return (e - 1.0f) / (e + 1.0f);
}

// A fragment for v_wmma_f32_16x16x32_bf16 from LDS h (row-major [BT][PAD] bf16):
// lane l: M = mrow + (l&15); elements 0-15 = K kb..kb+15, kb = k0 + (l>>4)*16
__device__ __forceinline__ v16bf load_a(const __bf16* A, int mrow, int k0, int lane) {
    const __bf16* p = A + (size_t)(mrow + (lane & 15)) * PAD + k0 + ((lane >> 4) << 4);
    v8bf lo = *(const v8bf*)p;
    v8bf hi = *(const v8bf*)(p + 8);
    return __builtin_shufflevector(lo, hi, 0,1,2,3,4,5,6,7,8,9,10,11,12,13,14,15);
}

// Accumulate gates += A(chunks of K=256 in LDS) x W^T  where W is row-major [1024][Kt] bf16.
// B fragment: lane l holds column n = gate*256+hbase+sub*16+(l&15),
// contiguous K run of 16 at (l>>4)*16 -> one 32-byte global load.
template <int NCH>
__device__ __forceinline__ void mm_acc(v8f acc[2][8], const __bf16* const* Achunks,
                                       const __bf16* __restrict__ W, int Kt,
                                       int hbase, int lane) {
#pragma unroll
    for (int c = 0; c < NCH; ++c) {
        const __bf16* A = Achunks[c];
#pragma unroll
        for (int ks = 0; ks < 8; ++ks) {
            const int k0 = ks * 32;
            v16bf a0 = load_a(A, 0,  k0, lane);
            v16bf a1 = load_a(A, 16, k0, lane);
#pragma unroll
            for (int ct = 0; ct < 8; ++ct) {
                const int gate = ct >> 1, sub = ct & 1;
                const int n = gate * 256 + hbase + sub * 16 + (lane & 15);
                const __bf16* bp = W + (size_t)n * Kt + c * 256 + k0 + ((lane >> 4) << 4);
                v16bf b = *(const v16bf*)bp;
                acc[0][ct] = __builtin_amdgcn_wmma_f32_16x16x32_bf16(
                    false, a0, false, b, (short)0, acc[0][ct], false, false);
                acc[1][ct] = __builtin_amdgcn_wmma_f32_16x16x32_bf16(
                    false, a1, false, b, (short)0, acc[1][ct], false, false);
            }
        }
    }
}

// bias add (+ optional one-hot input gather) + LSTM elementwise + write h (LDS) (+ optional out)
template <bool GATHER, bool WRITE_OUT>
__device__ __forceinline__ void finish_cell(
    v8f acc[2][8], float (*cst)[2][8], __bf16* __restrict__ hdst,
    const float* __restrict__ bias, const float* __restrict__ xg,
    const int posv[2][8], float* __restrict__ outp, int hbase, int lane) {
#pragma unroll
    for (int mt = 0; mt < 2; ++mt) {
#pragma unroll
        for (int sub = 0; sub < 2; ++sub) {
            const int col = hbase + sub * 16 + (lane & 15);
            float bi = bias[0 * H + col];
            float bf = bias[1 * H + col];
            float bg = bias[2 * H + col];
            float bo = bias[3 * H + col];
#pragma unroll
            for (int r = 0; r < 8; ++r) {
                const int m = mt * 16 + r + ((lane >> 4) << 3);
                float gi = acc[mt][0 + sub][r] + bi;
                float gf = acc[mt][2 + sub][r] + bf;
                float gg = acc[mt][4 + sub][r] + bg;
                float go = acc[mt][6 + sub][r] + bo;
                if (GATHER) {
                    int pos = posv[mt][r];
                    if (pos >= 0) {
                        const float* xrow = xg + (size_t)pos * G4;
                        gi += xrow[0 * H + col];
                        gf += xrow[1 * H + col];
                        gg += xrow[2 * H + col];
                        go += xrow[3 * H + col];
                    }
                }
                float c  = cst[mt][sub][r];
                float c2 = sigm(gf) * c + sigm(gi) * ftanh(gg);
                float h2 = sigm(go) * ftanh(c2);
                cst[mt][sub][r] = c2;
                hdst[(size_t)m * PAD + col] = (__bf16)h2;
                if (WRITE_OUT) outp[(size_t)m * (SEQL * 2 * H) + col] = h2;
            }
        }
    }
}

__global__ __launch_bounds__(256) void k_lstm(
    const int* __restrict__ data, const float* __restrict__ h0in, const float* __restrict__ c0in,
    const float* __restrict__ wih0T, const float* __restrict__ biasW,
    const __bf16* __restrict__ whh0, const __bf16* __restrict__ wih1,
    const __bf16* __restrict__ whh1, float* __restrict__ out) {
    // h buffers, bf16, cell order: 0=l0 dir0, 1=l0 dir1, 2=l1 dir0, 3=l1 dir1
    __shared__ __bf16 hb[4][BT * PAD];

    const int tid  = threadIdx.x;
    const int lane = tid & 31;
    const int wave = tid >> 5;
    const int hbase = wave * 32;       // this wave's hidden-unit slice [hbase, hbase+32)
    const int b0 = blockIdx.x * BT;

    // init LDS h state (fp32 -> bf16)
    for (int e = tid; e < 4 * BT * H; e += 256) {
        int cell = e / (BT * H);
        int rem  = e - cell * (BT * H);
        int m = rem / H, col = rem % H;
        hb[cell][m * PAD + col] =
            (__bf16)h0in[(size_t)cell * B_TOT * H + (size_t)(b0 + m) * H + col];
    }

    // init register-resident c state in the WMMA C/D layout
    float cst[4][2][2][8];
#pragma unroll
    for (int cell = 0; cell < 4; ++cell)
#pragma unroll
        for (int mt = 0; mt < 2; ++mt)
#pragma unroll
            for (int sub = 0; sub < 2; ++sub)
#pragma unroll
                for (int r = 0; r < 8; ++r) {
                    int m   = mt * 16 + r + ((lane >> 4) << 3);
                    int col = hbase + sub * 16 + (lane & 15);
                    cst[cell][mt][sub][r] =
                        c0in[(size_t)cell * B_TOT * H + (size_t)(b0 + m) * H + col];
                }
    __syncthreads();

    const v8f vzero = {0.f, 0.f, 0.f, 0.f, 0.f, 0.f, 0.f, 0.f};

    for (int t = 0; t < SEQL; ++t) {
        // board positions for this tile's batch rows (shared by both layer-0 dirs)
        int posv[2][8];
#pragma unroll
        for (int mt = 0; mt < 2; ++mt)
#pragma unroll
            for (int r = 0; r < 8; ++r) {
                int m = mt * 16 + r + ((lane >> 4) << 3);
                posv[mt][r] = data[(size_t)(b0 + m) * SEQL + t];
            }

        // ---- layer 0, dirs 0 and 1 (input contribution = column gather of Wih0) ----
        for (int d = 0; d < 2; ++d) {
            v8f acc[2][8];
#pragma unroll
            for (int mt = 0; mt < 2; ++mt)
#pragma unroll
                for (int ct = 0; ct < 8; ++ct) acc[mt][ct] = vzero;

            const __bf16* Ach[1] = { hb[d] };
            mm_acc<1>(acc, Ach, whh0 + (size_t)d * G4 * H, H, hbase, lane);
            __syncthreads();  // all waves finished reading old hb[d]
            finish_cell<true, false>(acc, cst[d], hb[d],
                                     biasW + (size_t)d * G4,
                                     wih0T + (size_t)d * BOARD * G4,
                                     posv, nullptr, hbase, lane);
            __syncthreads();  // new hb[d] visible
        }

        // ---- layer 1, dirs 0 and 1 (input = concat(h0f, h0b)) ----
        for (int d = 0; d < 2; ++d) {
            v8f acc[2][8];
#pragma unroll
            for (int mt = 0; mt < 2; ++mt)
#pragma unroll
                for (int ct = 0; ct < 8; ++ct) acc[mt][ct] = vzero;

            const __bf16* Ax[2] = { hb[0], hb[1] };
            mm_acc<2>(acc, Ax, wih1 + (size_t)d * G4 * (2 * H), 2 * H, hbase, lane);
            const __bf16* Ah[1] = { hb[2 + d] };
            mm_acc<1>(acc, Ah, whh1 + (size_t)d * G4 * H, H, hbase, lane);
            __syncthreads();  // all waves finished reading old hb[2+d]
            finish_cell<false, true>(acc, cst[2 + d], hb[2 + d],
                                     biasW + (size_t)(2 + d) * G4,
                                     nullptr, posv,
                                     out + (size_t)b0 * SEQL * 2 * H + (size_t)t * 2 * H + d * H,
                                     hbase, lane);
            __syncthreads();  // new hb[2+d] visible
        }
    }
}

// ---------------- launch ----------------

extern "C" void kernel_launch(void* const* d_in, const int* in_sizes, int n_in,
                              void* d_out, int out_size, void* d_ws, size_t ws_size,
                              hipStream_t stream) {
    const int*   data = (const int*)  d_in[0];
    const float* h0   = (const float*)d_in[1];
    const float* c0   = (const float*)d_in[2];
    const float* Wih0 = (const float*)d_in[3];
    const float* Whh0 = (const float*)d_in[4];
    const float* bih0 = (const float*)d_in[5];
    const float* bhh0 = (const float*)d_in[6];
    const float* Wih1 = (const float*)d_in[7];
    const float* Whh1 = (const float*)d_in[8];
    const float* bih1 = (const float*)d_in[9];
    const float* bhh1 = (const float*)d_in[10];
    float* out = (float*)d_out;

    // workspace carve-out (256B aligned sections)
    char* ws = (char*)d_ws;
    size_t off = 0;
    auto carve = [&](size_t bytes) -> void* {
        void* p = ws + off;
        off = (off + bytes + 255) & ~(size_t)255;
        return p;
    };
    const int nWhh0 = 2 * G4 * H;        // 524288
    const int nWih1 = 2 * G4 * 2 * H;    // 1048576
    const int nWhh1 = 2 * G4 * H;        // 524288
    __bf16* whh0b = (__bf16*)carve((size_t)nWhh0 * 2);
    __bf16* wih1b = (__bf16*)carve((size_t)nWih1 * 2);
    __bf16* whh1b = (__bf16*)carve((size_t)nWhh1 * 2);
    float*  wih0T = (float*) carve((size_t)2 * BOARD * G4 * 4);
    float*  biasW = (float*) carve((size_t)4 * G4 * 4);

    k_f32_to_bf16<<<(nWhh0 + 255) / 256, 256, 0, stream>>>(Whh0, whh0b, nWhh0);
    k_f32_to_bf16<<<(nWih1 + 255) / 256, 256, 0, stream>>>(Wih1, wih1b, nWih1);
    k_f32_to_bf16<<<(nWhh1 + 255) / 256, 256, 0, stream>>>(Whh1, whh1b, nWhh1);
    k_transpose_wih0<<<(2 * BOARD * G4 + 255) / 256, 256, 0, stream>>>(Wih0, wih0T);
    k_bias<<<(4 * G4 + 255) / 256, 256, 0, stream>>>(bih0, bhh0, bih1, bhh1, biasW);

    k_lstm<<<B_TOT / BT, 256, 0, stream>>>(data, h0, c0, wih0T, biasW,
                                           whh0b, wih1b, whh1b, out);
}
